// AFNONet_2250562863952
// MI455X (gfx1250) — matrix operand order repaired and checked by reading, not compiled, not run on hardware
//
#include <hip/hip_runtime.h>

typedef __attribute__((ext_vector_type(16))) __bf16 v16bf;
typedef __attribute__((ext_vector_type(8)))  __bf16 v8bf;
typedef __attribute__((ext_vector_type(8)))  float  v8f;

#define F_WF   1   // write fp32 output
#define F_WB   2   // write bf16 mirror output
#define F_GELU 4   // exact GELU epilogue
#define F_BETA 8   // accumulate Cin

#define BM 128
#define BN 128
#define BK 32
#define LDS_STRIDE 40   // elements; 80 bytes, multiple of 16B

// CDNA5 async global->LDS staging (ASYNCcnt-tracked), guarded so the file
// still compiles (with the round-1 vector-copy staging) if the toolchain
// lacks the builtins. Param type from hipcc diagnostic:
//   (v4i AS1*, v4i AS3*, Imm int offset, Imm int cpol)
#if __has_builtin(__builtin_amdgcn_global_load_async_to_lds_b128) && \
    __has_builtin(__builtin_amdgcn_s_wait_asynccnt)
#define USE_ASYNC 1
#define AS1 __attribute__((address_space(1)))
#define AS3 __attribute__((address_space(3)))
typedef int v4i_vs __attribute__((vector_size(16)));
// generic->AS1: full 64-bit address; generic->AS3: low 32 bits are the LDS offset
#define GPTR(p) ((AS1 v4i_vs*)(unsigned long long)(const void*)(p))
#define LPTR(p) ((AS3 v4i_vs*)(unsigned int)(unsigned long long)(const void*)(p))
#define ASYNC_WAIT() __builtin_amdgcn_s_wait_asynccnt(0)
#else
#define USE_ASYNC 0
#define ASYNC_WAIT()
#endif

// ---------------------------------------------------------------------------
// Generic strided-batched WMMA GEMM:  out = alpha*A@B [+ Cin] [+ bias] [GELU]
// A: bf16 MxK (lda), B: bf16 KxN (ldb), Cin/outF fp32, outB bf16 (ldc).
// Double-buffered LDS; A tile staged with GLOBAL_LOAD_ASYNC_TO_LDS_B128 when
// available (DMA for tile k+1 overlaps WMMAs on tile k).
// ---------------------------------------------------------------------------
__global__ __launch_bounds__(256)
void wmma_gemm_k(const __bf16* __restrict__ A, const __bf16* __restrict__ B,
                 const float* __restrict__ Cin, float* __restrict__ outF,
                 __bf16* __restrict__ outB, const float* __restrict__ bias,
                 int M, int N, int K, int lda, int ldb, int ldc,
                 long long bsA, long long bsB, long long bsC,
                 float alpha, int flags)
{
    __shared__ __bf16 As[2][BM * LDS_STRIDE];    // [m][k]
    __shared__ __bf16 Bs[2][BN * LDS_STRIDE];    // transposed: [n][k]

    const int tid  = threadIdx.x;
    const int lane = tid & 31;
    const int wid  = tid >> 5;
    const int bz   = blockIdx.z;

    const __bf16* Ab = A + (size_t)bz * bsA;
    const __bf16* Bb = B + (size_t)bz * bsB;
    const long long cOff = (long long)bz * bsC;

    const int m0 = blockIdx.y * BM;
    const int n0 = blockIdx.x * BN;

    const int wm = (wid >> 2) * 64;   // wave M offset: 0 / 64
    const int wn = (wid & 3) * 32;    // wave N offset: 0,32,64,96

    v8f acc[4][2];
#pragma unroll
    for (int i = 0; i < 4; ++i)
#pragma unroll
        for (int j = 0; j < 2; ++j) acc[i][j] = v8f{};

    // staging coordinates
    const int ra = tid >> 1;          // A: row within tile (0..127)
    const int ka = (tid & 1) * 16;    // A: k offset (0/16)
    const int rb = tid >> 3;          // B: k row within tile (0..31)
    const int nb = (tid & 7) * 16;    // B: n offset (0..112)

    const int laneM = lane & 15;
    const int kb = (lane < 16) ? 0 : 8;        // A-frag K base
    const int kh = (lane >> 4) * 16;           // B-frag K half

    // ---- tile staging (issue only; completion enforced by caller) ----
    auto stageA = [&](int k0s, int st) {
        const int gm = m0 + ra;
        const bool fast = (gm < M) && (k0s + ka + 16 <= K) && ((lda & 7) == 0);
        if (fast) {
            const __bf16* g = Ab + (size_t)gm * lda + k0s + ka;
            __bf16* l = &As[st][ra * LDS_STRIDE + ka];
#if USE_ASYNC
            __builtin_amdgcn_global_load_async_to_lds_b128(GPTR(g), LPTR(l), 0, 0);
            __builtin_amdgcn_global_load_async_to_lds_b128(GPTR(g), LPTR(l), 16, 0);
#else
            const int4* src = (const int4*)g;
            int4* dst = (int4*)l;
            dst[0] = src[0];
            dst[1] = src[1];
#endif
        } else {
            for (int e = 0; e < 16; ++e) {
                const int gk = k0s + ka + e;
                __bf16 v = (__bf16)0.0f;
                if (gm < M && gk < K) v = Ab[(size_t)gm * lda + gk];
                As[st][ra * LDS_STRIDE + ka + e] = v;
            }
        }
    };
    auto stageB = [&](int k0s, int st) {
        const int gk = k0s + rb;
        union { int4 v[2]; __bf16 h[16]; } t;
        const bool fast = (gk < K) && (n0 + nb + 16 <= N) && ((ldb & 7) == 0);
        if (fast) {
            const int4* src = (const int4*)(Bb + (size_t)gk * ldb + n0 + nb);
            t.v[0] = src[0];
            t.v[1] = src[1];
        } else {
            for (int e = 0; e < 16; ++e) {
                const int gn = n0 + nb + e;
                t.h[e] = (gk < K && gn < N) ? Bb[(size_t)gk * ldb + gn]
                                            : (__bf16)0.0f;
            }
        }
#pragma unroll
        for (int e = 0; e < 16; ++e) Bs[st][(nb + e) * LDS_STRIDE + rb] = t.h[e];
    };

    stageA(0, 0);
    stageB(0, 0);
    ASYNC_WAIT();
    __syncthreads();

    int cb = 0;
    for (int k0 = 0; k0 < K; k0 += BK) {
        // issue DMA / staging for the next tile into the other buffer
        if (k0 + BK < K) {
            stageA(k0 + BK, cb ^ 1);
            stageB(k0 + BK, cb ^ 1);
        }
        if (k0 + 2 * BK < K)  // prefetch the B tile after next
            __builtin_prefetch(Bb + (size_t)(k0 + 2 * BK + rb) * ldb + n0 + nb, 0, 1);

        // ---- load fragments per ISA layout and issue 8 WMMAs ----
        const __bf16* Ac = As[cb];
        const __bf16* Bc = Bs[cb];
        v16bf af[4];
        v16bf bfr[2];
#pragma unroll
        for (int i = 0; i < 4; ++i) {
            const int row = wm + i * 16 + laneM;
            v8bf lo = *(const v8bf*)(Ac + row * LDS_STRIDE + kb);
            v8bf hi = *(const v8bf*)(Ac + row * LDS_STRIDE + kb + 16);
            af[i] = __builtin_shufflevector(lo, hi, 0,1,2,3,4,5,6,7,8,9,10,11,12,13,14,15);
        }
#pragma unroll
        for (int j = 0; j < 2; ++j) {
            const int col = wn + j * 16 + laneM;
            v8bf lo = *(const v8bf*)(Bc + col * LDS_STRIDE + kb);
            v8bf hi = *(const v8bf*)(Bc + col * LDS_STRIDE + kb + 16);
            bfr[j] = __builtin_shufflevector(lo, hi, 0,1,2,3,4,5,6,7,8,9,10,11,12,13,14,15);
        }
#pragma unroll
        for (int i = 0; i < 4; ++i)
#pragma unroll
            for (int j = 0; j < 2; ++j)
                acc[i][j] = __builtin_amdgcn_wmma_f32_16x16x32_bf16(
                    false, af[i], false, bfr[j], (short)0, acc[i][j], false, false);

        ASYNC_WAIT();        // next-tile DMA complete before the barrier
        __syncthreads();     // orders DS stores + this tile's reads
        cb ^= 1;
    }

    // ---- epilogue: C/D fragment layout: lane n = lane&15, row = r + 8*(lane>>4)
    const int rowHi = (lane >> 4) << 3;
#pragma unroll
    for (int i = 0; i < 4; ++i) {
#pragma unroll
        for (int j = 0; j < 2; ++j) {
            const int gn = n0 + wn + j * 16 + laneM;
#pragma unroll
            for (int r = 0; r < 8; ++r) {
                const int gm = m0 + wm + i * 16 + r + rowHi;
                if (gm < M && gn < N) {
                    const long long idx = cOff + (long long)gm * ldc + gn;
                    float v = alpha * acc[i][j][r];
                    if (flags & F_BETA) v += Cin[idx];
                    if (bias)           v += bias[gn];
                    if (flags & F_GELU) v = 0.5f * v * (1.0f + erff(v * 0.70710678118654752f));
                    if (flags & F_WF)   outF[idx] = v;
                    if (flags & F_WB)   outB[idx] = (__bf16)v;
                }
            }
        }
    }
}

// ---------------------------------------------------------------------------
// LayerNorm over last dim C; writes fp32 + bf16 mirrors.
// ---------------------------------------------------------------------------
__global__ __launch_bounds__(256)
void layernorm_k(const float* __restrict__ x, const float* __restrict__ g,
                 const float* __restrict__ b, float* __restrict__ of,
                 __bf16* __restrict__ ob, int C)
{
    __shared__ float red[256];
    const int t = blockIdx.x;
    const float* xr = x + (size_t)t * C;

    float s = 0.0f;
    for (int i = threadIdx.x; i < C; i += 256) s += xr[i];
    red[threadIdx.x] = s; __syncthreads();
    for (int o = 128; o > 0; o >>= 1) {
        if (threadIdx.x < o) red[threadIdx.x] += red[threadIdx.x + o];
        __syncthreads();
    }
    const float mean = red[0] / (float)C;
    __syncthreads();

    float vs = 0.0f;
    for (int i = threadIdx.x; i < C; i += 256) { float d = xr[i] - mean; vs += d * d; }
    red[threadIdx.x] = vs; __syncthreads();
    for (int o = 128; o > 0; o >>= 1) {
        if (threadIdx.x < o) red[threadIdx.x] += red[threadIdx.x + o];
        __syncthreads();
    }
    const float rstd = rsqrtf(red[0] / (float)C + 1e-5f);

    for (int i = threadIdx.x; i < C; i += 256) {
        const float v = (xr[i] - mean) * rstd * g[i] + b[i];
        of[(size_t)t * C + i] = v;
        ob[(size_t)t * C + i] = (__bf16)v;
    }
}

// ---------------------------------------------------------------------------
// AFNO block-diagonal spectral MLP + soft-threshold for one (h,w) site.
// Mirrors the reference exactly (including o2n's use of o2k and Xnk from the
// spatial-domain roll/flip of the LN output).
// ---------------------------------------------------------------------------
__global__ __launch_bounds__(256)
void afno_block_k(const float* __restrict__ Xk, const float* __restrict__ lnf,
                  const float* __restrict__ w1, const float* __restrict__ b1,
                  const float* __restrict__ w2, const float* __restrict__ b2,
                  __bf16* __restrict__ Y)
{
    __shared__ float a_s[768], n_s[768], o1k[768], o1n[768], o2k[768];
    const int s = blockIdx.x;            // 0..1034
    const int h = s / 23, w = s % 23;
    const int hh = (45 - h) % 45, ww = (90 - w) % 90;
    const float* ar = Xk + (size_t)s * 768;
    const float* nr = lnf + ((size_t)hh * 90 + ww) * 768;
    for (int i = threadIdx.x; i < 768; i += 256) { a_s[i] = ar[i]; n_s[i] = nr[i]; }
    __syncthreads();

    for (int o = threadIdx.x; o < 768; o += 256) {
        const int nb = o / 96, oo = o % 96;
        const float* w10 = w1 + ((size_t)nb) * 96 * 96;         // w1[0][nb]
        const float* w11 = w1 + ((size_t)(8 + nb)) * 96 * 96;   // w1[1][nb]
        const float* av = a_s + nb * 96;
        const float* nv = n_s + nb * 96;
        float sk = 0.0f, sn = 0.0f;
        for (int i = 0; i < 96; ++i) {
            const float wa = w10[i * 96 + oo], wbv = w11[i * 96 + oo];
            const float wp = wa + wbv, wm = wa - wbv;
            sk += av[i] * wp + nv[i] * wm;
            sn += nv[i] * wp + av[i] * wm;
        }
        const float v1 = 0.5f * sk + b1[nb * 96 + oo];
        const float v2 = 0.5f * sn + b1[8 * 96 + nb * 96 + oo];
        o1k[o] = v1 > 0.0f ? v1 : 0.0f;
        o1n[o] = v2 > 0.0f ? v2 : 0.0f;
    }
    __syncthreads();

    for (int o = threadIdx.x; o < 768; o += 256) {
        const int nb = o / 96, oo = o % 96;
        const float* w20 = w2 + ((size_t)nb) * 96 * 96;
        const float* w21 = w2 + ((size_t)(8 + nb)) * 96 * 96;
        const float* k1 = o1k + nb * 96;
        const float* n1 = o1n + nb * 96;
        float sk = 0.0f;
        for (int i = 0; i < 96; ++i) {
            const float wa = w20[i * 96 + oo], wbv = w21[i * 96 + oo];
            sk += k1[i] * (wa + wbv) + n1[i] * (wa - wbv);
        }
        o2k[o] = 0.5f * sk + b2[nb * 96 + oo];
    }
    __syncthreads();

    for (int o = threadIdx.x; o < 768; o += 256) {
        const int nb = o / 96, oo = o % 96;
        const float* w20 = w2 + ((size_t)nb) * 96 * 96;
        const float* w21 = w2 + ((size_t)(8 + nb)) * 96 * 96;
        const float* n1 = o1n + nb * 96;
        const float* k2 = o2k + nb * 96;
        float sn = 0.0f;
        for (int i = 0; i < 96; ++i) {
            const float wa = w20[i * 96 + oo], wbv = w21[i * 96 + oo];
            sn += n1[i] * (wa + wbv) + k2[i] * (wa - wbv);
        }
        const float o2nv = 0.5f * sn + b2[8 * 96 + nb * 96 + oo];
        float y = o2k[o] + o2nv;
        const float ay = fabsf(y) - 0.01f;
        y = (y > 0.0f ? 1.0f : (y < 0.0f ? -1.0f : 0.0f)) * (ay > 0.0f ? ay : 0.0f);
        Y[(size_t)s * 768 + o] = (__bf16)y;
    }
}

// ---------------------------------------------------------------------------
// helpers: im2col, weight transpose+convert, DHT bases, residual add, scatter
// ---------------------------------------------------------------------------
__global__ void im2col_k(const float* __restrict__ x, __bf16* __restrict__ A)
{
    const int idx = blockIdx.x * 256 + threadIdx.x;
    if (idx >= 4050 * 5120) return;
    const int t = idx / 5120, col = idx % 5120;
    const int gh = t / 90, gw = t % 90;
    const int c = col / 256, rem = col % 256, ph = rem / 16, pw = rem % 16;
    A[idx] = (__bf16)x[((size_t)c * 720 + gh * 16 + ph) * 1440 + gw * 16 + pw];
}

__global__ void transpose_conv_k(const float* __restrict__ src, __bf16* __restrict__ dst,
                                 int R, int C, long long sStride, long long dStride)
{
    __shared__ float t[32][33];
    const int bz = blockIdx.z;
    const float* s = src + (size_t)bz * sStride;
    __bf16* d = dst + (size_t)bz * dStride;
    const int r0 = blockIdx.y * 32, c0 = blockIdx.x * 32;
    const int r = r0 + threadIdx.y, c = c0 + threadIdx.x;
    if (r < R && c < C) t[threadIdx.y][threadIdx.x] = s[(size_t)r * C + c];
    __syncthreads();
    const int rr = c0 + threadIdx.y, cc = r0 + threadIdx.x;   // dst is C x R
    if (rr < C && cc < R) d[(size_t)rr * R + cc] = (__bf16)t[threadIdx.x][threadIdx.y];
}

__global__ void dht_basis_k(__bf16* __restrict__ Cm, __bf16* __restrict__ Sm, int N)
{
    const int idx = blockIdx.x * 256 + threadIdx.x;
    if (idx >= N * N) return;
    const int i = idx / N, j = idx % N;
    const double ang = 6.283185307179586476 * (double)(((long long)i * j) % N) / (double)N;
    Cm[idx] = (__bf16)(float)cos(ang);
    Sm[idx] = (__bf16)(float)sin(ang);
}

__global__ void dht_basis_comb_k(__bf16* __restrict__ CmS, __bf16* __restrict__ CpS, int N)
{
    const int idx = blockIdx.x * 256 + threadIdx.x;
    if (idx >= N * N) return;
    const int i = idx / N, j = idx % N;
    const double ang = 6.283185307179586476 * (double)(((long long)i * j) % N) / (double)N;
    const float c = (float)cos(ang), s = (float)sin(ang);
    CmS[idx] = (__bf16)(c - s);
    CpS[idx] = (__bf16)(c + s);
}

__global__ void add3_k(float* __restrict__ cur, const float* __restrict__ a,
                       const float* __restrict__ b, int n)
{
    const int i = blockIdx.x * 256 + threadIdx.x;
    if (i < n) cur[i] = cur[i] + a[i] + b[i];
}

__global__ void head_scatter_k(const float* __restrict__ ho, float* __restrict__ out)
{
    const int idx = blockIdx.x * 256 + threadIdx.x;
    if (idx >= 20 * 720 * 1440) return;
    const int ch = idx / (720 * 1440);
    const int rem = idx % (720 * 1440);
    const int y = rem / 1440, xx = rem % 1440;
    const int gh = y / 16, ph = y % 16, gw = xx / 16, pw = xx % 16;
    out[idx] = ho[((size_t)(gh * 90 + gw)) * 5120 + ch * 256 + ph * 16 + pw];
}

// ---------------------------------------------------------------------------
// host side
// ---------------------------------------------------------------------------
static inline void gemm(hipStream_t st, const __bf16* A, const __bf16* B,
                        const float* Cin, float* oF, __bf16* oB, const float* bias,
                        int M, int N, int K, int lda, int ldb, int ldc,
                        long long bsA, long long bsB, long long bsC, int batch,
                        float alpha, int flags)
{
    dim3 g((N + BN - 1) / BN, (M + BM - 1) / BM, batch);
    wmma_gemm_k<<<g, dim3(256), 0, st>>>(A, B, Cin, oF, oB, bias, M, N, K,
                                         lda, ldb, ldc, bsA, bsB, bsC, alpha, flags);
}

extern "C" void kernel_launch(void* const* d_in, const int* in_sizes, int n_in,
                              void* d_out, int out_size, void* d_ws, size_t ws_size,
                              hipStream_t stream)
{
    const float* x        = (const float*)d_in[0];
    const float* patch_w  = (const float*)d_in[1];
    const float* patch_b  = (const float*)d_in[2];
    const float* pos_emb  = (const float*)d_in[3];
    const float* norm1_w  = (const float*)d_in[4];
    const float* norm1_b  = (const float*)d_in[5];
    const float* w1       = (const float*)d_in[6];
    const float* b1       = (const float*)d_in[7];
    const float* w2       = (const float*)d_in[8];
    const float* b2       = (const float*)d_in[9];
    const float* norm2_w  = (const float*)d_in[10];
    const float* norm2_b  = (const float*)d_in[11];
    const float* fc1_w    = (const float*)d_in[12];
    const float* fc1_b    = (const float*)d_in[13];
    const float* fc2_w    = (const float*)d_in[14];
    const float* fc2_b    = (const float*)d_in[15];
    const float* head_w   = (const float*)d_in[16];
    float* out = (float*)d_out;

    const int NT = 4050, E = 768, HID = 3072, KP = 5120, NO = 5120;
    const int H = 45, W = 90, KM = 23, S1 = H * KM;  // 1035
    const long long NE = (long long)NT * E;          // 4050*768

    // bump allocator over d_ws
    char* p = (char*)d_ws;
    auto alloc = [&](size_t bytes) -> void* {
        void* r = (void*)p;
        p += (bytes + 255) & ~(size_t)255;
        return r;
    };
    // bf16 weights / bases
    __bf16* patchT = (__bf16*)alloc((size_t)KP * E * 2);
    __bf16* fc1T   = (__bf16*)alloc((size_t)12 * E * HID * 2);
    __bf16* fc2T   = (__bf16*)alloc((size_t)12 * HID * E * 2);
    __bf16* headT  = (__bf16*)alloc((size_t)E * NO * 2);
    __bf16* CH  = (__bf16*)alloc((size_t)H * H * 2);
    __bf16* SH  = (__bf16*)alloc((size_t)H * H * 2);
    __bf16* CW  = (__bf16*)alloc((size_t)W * W * 2);
    __bf16* SW  = (__bf16*)alloc((size_t)W * W * 2);
    __bf16* CmS = (__bf16*)alloc((size_t)E * E * 2);
    __bf16* CpS = (__bf16*)alloc((size_t)E * E * 2);
    // activations
    __bf16* Aim  = (__bf16*)alloc((size_t)NT * KP * 2);
    float*  cur  = (float*)alloc((size_t)NE * 4);
    __bf16* curb = (__bf16*)alloc((size_t)NE * 2);
    float*  lnf  = (float*)alloc((size_t)NE * 4);
    __bf16* lnb  = (__bf16*)alloc((size_t)NE * 2);
    __bf16* Re0b = (__bf16*)alloc((size_t)NE * 2);
    __bf16* Im0b = (__bf16*)alloc((size_t)NE * 2);
    float*  Re1f = (float*)alloc((size_t)S1 * E * 4);
    float*  Im1f = (float*)alloc((size_t)S1 * E * 4);
    __bf16* Re1b = (__bf16*)alloc((size_t)S1 * E * 2);
    __bf16* Im1b = (__bf16*)alloc((size_t)S1 * E * 2);
    float*  Xk   = (float*)alloc((size_t)S1 * E * 4);
    __bf16* Ybf  = (__bf16*)alloc((size_t)S1 * E * 2);
    __bf16* Rhb  = (__bf16*)alloc((size_t)S1 * E * 2);
    __bf16* Ihb  = (__bf16*)alloc((size_t)S1 * E * 2);
    float*  Rwf  = (float*)alloc((size_t)NE * 4);
    float*  Iwf  = (float*)alloc((size_t)NE * 4);
    __bf16* Rwb  = (__bf16*)alloc((size_t)NE * 2);
    __bf16* Iwb  = (__bf16*)alloc((size_t)NE * 2);
    float*  tmp  = (float*)alloc((size_t)NE * 4);
    __bf16* midb = (__bf16*)alloc((size_t)NT * HID * 2);
    float*  hout = (float*)alloc((size_t)NT * NO * 4);

    // ---- one-time (per launch) prep: bases + weight transpose/convert ----
    dht_basis_k<<<(H * H + 255) / 256, 256, 0, stream>>>(CH, SH, H);
    dht_basis_k<<<(W * W + 255) / 256, 256, 0, stream>>>(CW, SW, W);
    dht_basis_comb_k<<<(E * E + 255) / 256, 256, 0, stream>>>(CmS, CpS, E);

    {
        dim3 b(32, 32);
        transpose_conv_k<<<dim3((KP + 31) / 32, (E + 31) / 32, 1), b, 0, stream>>>(
            patch_w, patchT, E, KP, 0, 0);                       // (768x5120) -> (5120x768)
        transpose_conv_k<<<dim3((E + 31) / 32, (HID + 31) / 32, 12), b, 0, stream>>>(
            fc1_w, fc1T, HID, E, (long long)HID * E, (long long)E * HID);
        transpose_conv_k<<<dim3((HID + 31) / 32, (E + 31) / 32, 12), b, 0, stream>>>(
            fc2_w, fc2T, E, HID, (long long)E * HID, (long long)HID * E);
        transpose_conv_k<<<dim3((E + 31) / 32, (NO + 31) / 32, 1), b, 0, stream>>>(
            head_w, headT, NO, E, 0, 0);                         // (5120x768) -> (768x5120)
    }

    // ---- patch embedding: cur = im2col(x) @ patchT + patch_b + pos_embed ----
    im2col_k<<<(NT * KP + 255) / 256, 256, 0, stream>>>(x, Aim);
    gemm(stream, Aim, patchT, pos_emb, cur, nullptr, patch_b,
         NT, E, KP, KP, E, E, 0, 0, 0, 1, 1.0f, F_WF | F_BETA);

    const float invSize = 1.0f / (float)(45 * 90 * 768);

    for (int l = 0; l < 12; ++l) {
        const __bf16* f1T = fc1T + (size_t)l * E * HID;
        const __bf16* f2T = fc2T + (size_t)l * HID * E;

        layernorm_k<<<NT, 256, 0, stream>>>(cur, norm1_w + l * E, norm1_b + l * E,
                                            lnf, lnb, E);
        // forward DHT, axis H (real input -> Re/Im)
        gemm(stream, CH, lnb, nullptr, nullptr, Re0b, nullptr,
             H, W * E, H, H, W * E, W * E, 0, 0, 0, 1, 1.0f, F_WB);
        gemm(stream, SH, lnb, nullptr, nullptr, Im0b, nullptr,
             H, W * E, H, H, W * E, W * E, 0, 0, 0, 1, -1.0f, F_WB);
        // axis W (batched over the 45 H-frequencies, only k<23 outputs kept)
        gemm(stream, CW, Re0b, nullptr, Re1f, nullptr, nullptr,
             KM, E, W, W, E, E, 0, (long long)W * E, (long long)KM * E, H, 1.0f, F_WF);
        gemm(stream, SW, Im0b, Re1f, Re1f, Re1b, nullptr,
             KM, E, W, W, E, E, 0, (long long)W * E, (long long)KM * E, H, 1.0f,
             F_WF | F_WB | F_BETA);
        gemm(stream, CW, Im0b, nullptr, Im1f, nullptr, nullptr,
             KM, E, W, W, E, E, 0, (long long)W * E, (long long)KM * E, H, 1.0f, F_WF);
        gemm(stream, SW, Re0b, Im1f, Im1f, Im1b, nullptr,
             KM, E, W, W, E, E, 0, (long long)W * E, (long long)KM * E, H, -1.0f,
             F_WF | F_WB | F_BETA);
        // axis C combined: Xk = Re@(C-S) + Im@(C+S)
        gemm(stream, Re1b, CmS, nullptr, Xk, nullptr, nullptr,
             S1, E, E, E, E, E, 0, 0, 0, 1, 1.0f, F_WF);
        gemm(stream, Im1b, CpS, Xk, Xk, nullptr, nullptr,
             S1, E, E, E, E, E, 0, 0, 0, 1, 1.0f, F_WF | F_BETA);
        // spectral block MLP + soft-threshold
        afno_block_k<<<S1, 256, 0, stream>>>(Xk, lnf,
                                             w1 + (size_t)l * 2 * 8 * 96 * 96,
                                             b1 + (size_t)l * 2 * 8 * 96,
                                             w2 + (size_t)l * 2 * 8 * 96 * 96,
                                             b2 + (size_t)l * 2 * 8 * 96, Ybf);
        // inverse DHT, axis H (batched over 23 W-columns)
        gemm(stream, CH, Ybf, nullptr, nullptr, Rhb, nullptr,
             H, E, H, H, KM * E, KM * E, 0, E, E, KM, 1.0f, F_WB);
        gemm(stream, SH, Ybf, nullptr, nullptr, Ihb, nullptr,
             H, E, H, H, KM * E, KM * E, 0, E, E, KM, -1.0f, F_WB);
        // axis W (batched over 45 rows, K=23 nonzero spectrum)
        gemm(stream, CW, Rhb, nullptr, Rwf, nullptr, nullptr,
             W, E, KM, W, E, E, 0, (long long)KM * E, (long long)W * E, H, 1.0f, F_WF);
        gemm(stream, SW, Ihb, Rwf, Rwf, Rwb, nullptr,
             W, E, KM, W, E, E, 0, (long long)KM * E, (long long)W * E, H, 1.0f,
             F_WF | F_WB | F_BETA);
        gemm(stream, CW, Ihb, nullptr, Iwf, nullptr, nullptr,
             W, E, KM, W, E, E, 0, (long long)KM * E, (long long)W * E, H, 1.0f, F_WF);
        gemm(stream, SW, Rhb, Iwf, Iwf, Iwb, nullptr,
             W, E, KM, W, E, E, 0, (long long)KM * E, (long long)W * E, H, -1.0f,
             F_WF | F_WB | F_BETA);
        // axis C combined + 1/size scaling
        gemm(stream, Rwb, CmS, nullptr, tmp, nullptr, nullptr,
             NT, E, E, E, E, E, 0, 0, 0, 1, invSize, F_WF);
        gemm(stream, Iwb, CpS, tmp, tmp, nullptr, nullptr,
             NT, E, E, E, E, E, 0, 0, 0, 1, invSize, F_WF | F_BETA);
        // cur = dht_out + ln_out(bias) + residual(cur)
        add3_k<<<((int)NE + 255) / 256, 256, 0, stream>>>(cur, tmp, lnf, (int)NE);

        // MLP
        layernorm_k<<<NT, 256, 0, stream>>>(cur, norm2_w + l * E, norm2_b + l * E,
                                            lnf, lnb, E);
        gemm(stream, lnb, f1T, nullptr, nullptr, midb, fc1_b + l * HID,
             NT, HID, E, E, HID, HID, 0, 0, 0, 1, 1.0f, F_WB | F_GELU);
        gemm(stream, midb, f2T, cur, cur, curb, fc2_b + l * E,
             NT, E, HID, HID, E, E, 0, 0, 0, 1, 1.0f, F_WF | F_WB | F_BETA);
    }

    // ---- head + reassembly ----
    gemm(stream, curb, headT, nullptr, hout, nullptr, nullptr,
         NT, NO, E, E, NO, NO, 0, 0, 0, 1, 1.0f, F_WF);
    head_scatter_k<<<(20 * 720 * 1440 + 255) / 256, 256, 0, stream>>>(hout, out);
}
